// SecondOrderRNN_66838281060505
// MI455X (gfx1250) — matrix operand order repaired and checked by reading, not compiled, not run on hardware
//
#include <hip/hip_runtime.h>
#include <hip/hip_bf16.h>

// SecondOrderRNN on gfx1250 (MI455X).
// S=512,B=64,I=128,H=256,O=128.  Dominant op: per-step GEMM (64 x 33152) x (33152 x 256)
// in bf16 WMMA, fp32 accumulate.  A'=[A|U|V] packed once to bf16 in WMMA-B layout
// (17MB, L2 resident).  4 independent recurrence chains (16 batch rows each), 16 WGs per
// chain (one per 16-wide N tile), per-step sync via per-step arrive counters in d_ws.
// Key invariant: wave w's K-chunks (c = w mod 8) all use the SAME 32-wide h window
// (kb&255 == w*32), so the A-fragment h operands are hoisted to registers per step.
// Inner loop: depth-1 pipeline on BOTH the B-fragment global stream and the x LDS
// broadcast (round 3 showed a per-iteration s_wait_dscnt stall); unroll forced to 1
// to keep the live set small (round 2 showed unroll-4 + pipelining spills to scratch).

typedef __attribute__((ext_vector_type(16))) __bf16 v16bf;
typedef __attribute__((ext_vector_type(8)))  float  v8f;

constexpr int S = 512, B = 64, I = 128, H = 256, O = 128;
constexpr int KAUG  = I * H + I + H;   // 33152 = x-outer-h | x | h
constexpr int NCH   = KAUG / 32;       // 1036 K-chunks of 32
constexpr int NT_H  = H / 16;          // 16 N tiles for h_new
constexpr int NT_O  = O / 16;          // 8  N tiles for y
constexpr int WCH   = H / 32;          // 8  K-chunks for y GEMM
constexpr int NCHAIN = B / 16;         // 4 independent chains

// workspace layout (bytes)
constexpr size_t AP_ELEMS = (size_t)NT_H * NCH * 512;   // bf16 elems: 8,486,912
constexpr size_t WP_ELEMS = (size_t)NT_O * WCH * 512;   // bf16 elems: 32,768
constexpr size_t AP_BYTES = AP_ELEMS * 2;               // ~16.2 MB
constexpr size_t WP_BYTES = WP_ELEMS * 2;
constexpr size_t HB_BYTES = (size_t)2 * B * H * 4;      // double-buffered h state
constexpr size_t CNT_INTS = (size_t)NCHAIN * S;         // per-step arrive counters

__device__ __forceinline__ unsigned short f2bf_bits(float f) {
  union { float f; unsigned int u; } v; v.f = f;
  unsigned int u = v.u + 0x7FFFu + ((v.u >> 16) & 1u);  // round-to-nearest-even
  return (unsigned short)(u >> 16);
}

// ---- pack A' = [A | U | V] into bf16, WMMA B-operand layout --------------------
// layout: ap[nt][chunk][lane][e], lane&15 = N col, K = (e>>3)*16 + (lane>>4)*8 + (e&7)
__global__ void pack_A(const float* __restrict__ A, const float* __restrict__ U,
                       const float* __restrict__ V, unsigned short* __restrict__ ap) {
  size_t t = (size_t)blockIdx.x * 256 + threadIdx.x;
  if (t >= AP_ELEMS) return;
  int e    = (int)(t & 15);
  int lane = (int)((t >> 4) & 31);
  int c    = (int)((t >> 9) % NCH);
  int nt   = (int)((t >> 9) / NCH);
  int k = c * 32 + ((e >> 3) * 16 + (lane >> 4) * 8 + (e & 7));
  int n = nt * 16 + (lane & 15);
  float v;
  if (k < I * H)              v = A[(size_t)n * (I * H) + k];
  else if (k < I * H + I)     v = U[n * I + (k - I * H)];
  else                        v = V[n * H + (k - I * H - I)];
  ap[t] = f2bf_bits(v);
}

// ---- pack W (O,H) into bf16 WMMA B layout for y = h_new @ W^T ------------------
__global__ void pack_W(const float* __restrict__ W, unsigned short* __restrict__ wp) {
  size_t t = (size_t)blockIdx.x * 256 + threadIdx.x;
  if (t >= WP_ELEMS) return;
  int e    = (int)(t & 15);
  int lane = (int)((t >> 4) & 31);
  int c    = (int)((t >> 9) % WCH);
  int nt   = (int)((t >> 9) / WCH);
  int k = c * 32 + ((e >> 3) * 16 + (lane >> 4) * 8 + (e & 7));
  int n = nt * 16 + (lane & 15);
  wp[t] = f2bf_bits(W[(size_t)n * H + k]);
}

// ---- init: h_state <- hidden, zero arrive counters -----------------------------
__global__ void init_state(const float* __restrict__ hidden, float* __restrict__ hbuf,
                           int* __restrict__ cnt) {
  int t = blockIdx.x * 256 + threadIdx.x;
  if (t < B * H) hbuf[t] = hidden[t];
  else if (t < B * H + (int)CNT_INTS) cnt[t - B * H] = 0;
}

// A-fragment = xi * hoisted h window (16x32 A layout: K = (e>>3)*16 + half*8 + (e&7))
__device__ __forceinline__ v16bf mk_afrag(float xi, float4 h0, float4 h1,
                                          float4 h2, float4 h3) {
  v16bf a;
  a[0]  = (__bf16)(xi * h0.x); a[1]  = (__bf16)(xi * h0.y);
  a[2]  = (__bf16)(xi * h0.z); a[3]  = (__bf16)(xi * h0.w);
  a[4]  = (__bf16)(xi * h1.x); a[5]  = (__bf16)(xi * h1.y);
  a[6]  = (__bf16)(xi * h1.z); a[7]  = (__bf16)(xi * h1.w);
  a[8]  = (__bf16)(xi * h2.x); a[9]  = (__bf16)(xi * h2.y);
  a[10] = (__bf16)(xi * h2.z); a[11] = (__bf16)(xi * h2.w);
  a[12] = (__bf16)(xi * h3.x); a[13] = (__bf16)(xi * h3.y);
  a[14] = (__bf16)(xi * h3.z); a[15] = (__bf16)(xi * h3.w);
  return a;
}

// A-fragment straight from an fp32 LDS row; p = &row[chunkbase + half*8]
__device__ __forceinline__ v16bf mk_frag_lds(const float* p) {
  float4 a0 = *(const float4*)(p);      float4 a1 = *(const float4*)(p + 4);
  float4 b0 = *(const float4*)(p + 16); float4 b1 = *(const float4*)(p + 20);
  v16bf a;
  a[0]  = (__bf16)a0.x; a[1]  = (__bf16)a0.y; a[2]  = (__bf16)a0.z; a[3]  = (__bf16)a0.w;
  a[4]  = (__bf16)a1.x; a[5]  = (__bf16)a1.y; a[6]  = (__bf16)a1.z; a[7]  = (__bf16)a1.w;
  a[8]  = (__bf16)b0.x; a[9]  = (__bf16)b0.y; a[10] = (__bf16)b0.z; a[11] = (__bf16)b0.w;
  a[12] = (__bf16)b1.x; a[13] = (__bf16)b1.y; a[14] = (__bf16)b1.z; a[15] = (__bf16)b1.w;
  return a;
}

// ---- persistent per-chain RNN kernel -------------------------------------------
__global__ __launch_bounds__(256, 1) void rnn_persist(
    const float* __restrict__ xin, const float* __restrict__ bias,
    const float* __restrict__ cbias,
    const unsigned short* __restrict__ ap, const unsigned short* __restrict__ wp,
    float* __restrict__ hbuf, int* __restrict__ cnt,
    float* __restrict__ yout, float* __restrict__ hout) {
  __shared__ float x_lds[16 * I];     // 8 KB
  __shared__ float h_lds[16 * H];     // 16 KB
  __shared__ float acc_lds[256];      // 1 KB  (16x16 preact tile)

  const int t    = threadIdx.x;
  const int lane = t & 31;
  const int wave = t >> 5;
  const int half = lane >> 4;
  const int nn   = lane & 15;         // N col within tile
  const int m    = lane & 15;         // M row for A fragments
  const int chain = blockIdx.x >> 4;  // 0..3
  const int nt    = blockIdx.x & 15;  // 0..15 N-tile of H
  const int bbase = chain * 16;

  // preload h state (after 0 steps)
  for (int j = t; j < 16 * H; j += 256)
    h_lds[j] = hbuf[(size_t)(bbase + (j >> 8)) * H + (j & 255)];
  __syncthreads();

  const unsigned short* apbase = ap + (size_t)nt * NCH * 512;
  // this wave's fixed per-lane B pointer for chunk (wave + 8*i): stride 4096 elems
  const unsigned short* pb = apbase + ((size_t)wave * 32 + lane) * 16;

  for (int s = 0; s < S; ++s) {
    // stage x_s tile and seed accumulator with bias
    for (int j = t; j < 16 * I; j += 256)
      x_lds[j] = xin[((size_t)s * B + bbase + (j >> 7)) * I + (j & 127)];
    acc_lds[t] = bias[nt * 16 + (t & 15)];
    __syncthreads();

    // hoist this wave's fixed h window (kb&255 == wave*32 for every main chunk)
    const float* hwin = &h_lds[m * H + wave * 32 + half * 8];
    const float4 h0 = *(const float4*)(hwin);
    const float4 h1 = *(const float4*)(hwin + 4);
    const float4 h2 = *(const float4*)(hwin + 16);
    const float4 h3 = *(const float4*)(hwin + 20);

    v8f acc = {0.f, 0.f, 0.f, 0.f, 0.f, 0.f, 0.f, 0.f};

    // main region: 128 chunks/wave, i = c/8 indexes x columns.
    // depth-1 pipeline on both the B fragment (global) and x broadcast (LDS);
    // unroll 1 keeps the live set small (no spills).
    v16bf bcur = *(const v16bf*)(pb);
    float xcur = x_lds[m * I];
    #pragma unroll 1
    for (int i = 0; i < 127; ++i) {
      v16bf bnext = *(const v16bf*)(pb + (size_t)(i + 1) * 4096);
      const float xnext = x_lds[m * I + i + 1];
      __builtin_prefetch(pb + (size_t)(i + 8) * 4096, 0, 1);
      acc = __builtin_amdgcn_wmma_f32_16x16x32_bf16(
          false, mk_afrag(xcur, h0, h1, h2, h3), false, bcur, (short)0, acc,
          false, false);
      bcur = bnext;
      xcur = xnext;
    }
    acc = __builtin_amdgcn_wmma_f32_16x16x32_bf16(
        false, mk_afrag(xcur, h0, h1, h2, h3), false, bcur, (short)0, acc,
        false, false);

    // augmented tail, chunk-aligned, wave-uniform branches:
    // chunks 1024..1027 (x source) -> waves 0..3; chunks 1028..1035 (h source) -> all
    if (wave < 4) {
      v16bf bf = *(const v16bf*)(apbase + ((size_t)(1024 + wave) * 32 + lane) * 16);
      v16bf a  = mk_frag_lds(&x_lds[m * I + wave * 32 + half * 8]);
      acc = __builtin_amdgcn_wmma_f32_16x16x32_bf16(
          false, a, false, bf, (short)0, acc, false, false);
    }
    {
      v16bf bf = *(const v16bf*)(apbase + ((size_t)(1028 + wave) * 32 + lane) * 16);
      v16bf a  = mk_frag_lds(&h_lds[m * H + wave * 32 + half * 8]);
      acc = __builtin_amdgcn_wmma_f32_16x16x32_bf16(
          false, a, false, bf, (short)0, acc, false, false);
    }

    // cross-wave K reduction into LDS tile (C/D layout: row = r + half*8, col = nn)
    #pragma unroll
    for (int r = 0; r < 8; ++r)
      atomicAdd(&acc_lds[(r + half * 8) * 16 + nn], acc[r]);
    __syncthreads();

    // relu + publish h_new tile (double-buffered state + hidden_out)
    {
      float v = acc_lds[t];
      v = v > 0.f ? v : 0.f;
      const int mm = t >> 4, nc = t & 15;
      const size_t row = (size_t)bbase + mm;
      float* hnext = hbuf + (size_t)((s + 1) & 1) * B * H;
      hnext[row * H + nt * 16 + nc] = v;
      hout[((size_t)s * B + row) * H + nt * 16 + nc] = v;
    }
    __threadfence();
    __syncthreads();

    // chain-wide arrive/wait (16 WGs per chain, per-step counter)
    int* c0 = cnt + chain * S + s;
    if (t == 0) {
      __hip_atomic_fetch_add(c0, 1, __ATOMIC_RELEASE, __HIP_MEMORY_SCOPE_AGENT);
      while (__hip_atomic_load(c0, __ATOMIC_ACQUIRE, __HIP_MEMORY_SCOPE_AGENT) < 16)
        __builtin_amdgcn_s_sleep(2);
    }
    __syncthreads();

    // pull full h_new into LDS (it is next step's state)
    {
      const float* hnext = hbuf + (size_t)((s + 1) & 1) * B * H;
      for (int j = t; j < 16 * H; j += 256)
        h_lds[j] = hnext[(size_t)(bbase + (j >> 8)) * H + (j & 255)];
    }
    __syncthreads();

    // y_s = h_new @ W^T + c  (tiles 0..7, wave 0; wave-uniform branch)
    if (nt < NT_O && wave == 0) {
      v8f yacc = {0.f, 0.f, 0.f, 0.f, 0.f, 0.f, 0.f, 0.f};
      const unsigned short* wpb = wp + (size_t)nt * WCH * 512;
      #pragma unroll
      for (int c2 = 0; c2 < WCH; ++c2) {
        v16bf bfrag = *(const v16bf*)(wpb + ((size_t)c2 * 32 + lane) * 16);
        v16bf afrag = mk_frag_lds(&h_lds[m * H + c2 * 32 + half * 8]);
        yacc = __builtin_amdgcn_wmma_f32_16x16x32_bf16(
            false, afrag, false, bfrag, (short)0, yacc, false, false);
      }
      const float cb = cbias[nt * 16 + nn];
      #pragma unroll
      for (int r = 0; r < 8; ++r) {
        int mm = r + half * 8;
        yout[((size_t)s * B + bbase + mm) * O + nt * 16 + nn] = yacc[r] + cb;
      }
    }
  }
}

extern "C" void kernel_launch(void* const* d_in, const int* in_sizes, int n_in,
                              void* d_out, int out_size, void* d_ws, size_t ws_size,
                              hipStream_t stream) {
  const float* xin    = (const float*)d_in[0];  // (S,B,I)
  const float* hidden = (const float*)d_in[1];  // (B,H)
  const float* A      = (const float*)d_in[2];  // (H, I*H)
  const float* U      = (const float*)d_in[3];  // (H, I)
  const float* V      = (const float*)d_in[4];  // (H, H)
  const float* bias   = (const float*)d_in[5];  // (H,)
  const float* W      = (const float*)d_in[6];  // (O, H)
  const float* cbias  = (const float*)d_in[7];  // (O,)

  char* ws = (char*)d_ws;                       // needs ~17.2 MB
  unsigned short* ap = (unsigned short*)ws;
  unsigned short* wp = (unsigned short*)(ws + AP_BYTES);
  float* hbuf        = (float*)(ws + AP_BYTES + WP_BYTES);
  int* cnt           = (int*)(ws + AP_BYTES + WP_BYTES + HB_BYTES);

  float* yout = (float*)d_out;                  // (S,B,O)
  float* hout = yout + (size_t)S * B * O;       // (S,B,H)

  pack_A<<<(unsigned)((AP_ELEMS + 255) / 256), 256, 0, stream>>>(A, U, V, ap);
  pack_W<<<(unsigned)((WP_ELEMS + 255) / 256), 256, 0, stream>>>(W, wp);
  init_state<<<(unsigned)((B * H + CNT_INTS + 255) / 256), 256, 0, stream>>>(hidden, hbuf, cnt);
  rnn_persist<<<NCHAIN * NT_H, 256, 0, stream>>>(xin, bias, cbias, ap, wp,
                                                 hbuf, cnt, yout, hout);
}
